// ARVideoPatchTransformer_80436147519663
// MI455X (gfx1250) — matrix operand order repaired and checked by reading, not compiled, not run on hardware
//
#include <hip/hip_runtime.h>
#include <hip/hip_bf16.h>
#include <cstdint>
#include <cstddef>

// ---------------------------------------------------------------------------
// AR video patch transformer forward, CDNA5 (gfx1250), bf16 WMMA everywhere,
// async global->LDS tile staging via inline CDNA5 assembly (ASYNCcnt path).
// ---------------------------------------------------------------------------

typedef __bf16 bf16;
typedef __attribute__((ext_vector_type(16))) __bf16      v16bf;
typedef __attribute__((ext_vector_type(8)))  float       v8f;
typedef __attribute__((ext_vector_type(8)))  unsigned int u32x8;

// problem constants
constexpr int RES_ = 64;
constexpr int PP   = 8;
constexpr int CC   = 3;
constexpr int DD   = 512;
constexpr int NHD  = 8;
constexpr int NLAY = 8;
constexpr int TT   = 8;            // FIN + FOUT frames
constexpr int NP_  = 64;           // (RES/P)^2 patches per frame
constexpr int PD_  = CC * PP * PP; // 192
constexpr int HD_  = DD / NHD;     // 64
constexpr int RD_  = HD_ / 2;      // 32
constexpr int INNER_ = 1364;
constexpr int HALF_  = INNER_ / 2; // 682
constexpr int BB   = 32;
constexpr int LL   = TT * NP_;     // 512
constexpr int MM   = BB * LL;      // 16384 rows in the token matrix
constexpr float EPS_ = 1e-6f;

// ---------------------------------------------------------------------------
// CDNA5 async global->LDS copies (ASYNCcnt-tracked), via inline asm so we
// bypass the clang builtin's CUDA-address-space pointer types entirely.
// VDST operand = 32-bit LDS byte address (addr[31:0] of the generic pointer,
// per the ISA aperture rule); VADDR = 64-bit global address; GV mode (off).
// ---------------------------------------------------------------------------
#if defined(__AMDGCN__)
#define USE_ASYNC_LDS 1
#else
#define USE_ASYNC_LDS 0
#endif

__device__ inline void async_copy16(const bf16* g, bf16* l) {
#if USE_ASYNC_LDS
  const unsigned lds = (unsigned)(unsigned long long)l;
  asm volatile("global_load_async_to_lds_b128 %0, %1, off"
               :
               : "v"(lds), "v"(g)
               : "memory");
#else
  *(uint4*)l = *(const uint4*)g;
#endif
}

__device__ inline void wait_async_lds() {
#if USE_ASYNC_LDS
  asm volatile("s_wait_asynccnt 0x0" ::: "memory");
#endif
}

// ---------------------------------------------------------------------------
// WMMA helpers
// ---------------------------------------------------------------------------
__device__ inline v8f wmma_bf16(v16bf a, v16bf b, v8f c) {
  return __builtin_amdgcn_wmma_f32_16x16x32_bf16(
      /*neg_a=*/false, a, /*neg_b=*/false, b,
      /*c_mod=*/(short)0, c, /*reuse_a=*/false, /*reuse_b=*/false);
}

// A fragment (16x32, MxK): lane l<16 -> row M=l, K 0..15; lanes 16..31 hold
// K 16..31, values contiguous. `row` points at 32 contiguous bf16.
__device__ inline v16bf frag_a(const bf16* row, int half) {
  const unsigned int* r32 = (const unsigned int*)row;
  u32x8 t;
  const int o = half * 8;
  t[0] = r32[o + 0]; t[1] = r32[o + 1]; t[2] = r32[o + 2]; t[3] = r32[o + 3];
  t[4] = r32[o + 4]; t[5] = r32[o + 5]; t[6] = r32[o + 6]; t[7] = r32[o + 7];
  return __builtin_bit_cast(v16bf, t);
}

// B fragment (32x16, KxN): lane = column, lanes 0-15 K 0..15, lanes 16-31
// K 16..31, values contiguous.  `row` points at 32 contiguous bf16 (K-major).
__device__ inline v16bf frag_b(const bf16* row, int half) {
  const unsigned int* r32 = (const unsigned int*)row + half * 8;
  u32x8 t;
#pragma unroll
  for (int i = 0; i < 8; ++i) t[i] = r32[i];
  return __builtin_bit_cast(v16bf, t);
}

// ---------------------------------------------------------------------------
// Elementwise kernels
// ---------------------------------------------------------------------------
__global__ void k_f32_to_bf16(const float* __restrict__ in,
                              bf16* __restrict__ out, int n) {
  int i = blockIdx.x * blockDim.x + threadIdx.x;
  if (i < n) out[i] = (bf16)in[i];
}

// frames [B,T,C,RES,RES] -> bf16 token matrix [B*L, PD]
__global__ void k_patchify(const float* __restrict__ fr, bf16* __restrict__ A) {
  long long i = (long long)blockIdx.x * blockDim.x + threadIdx.x;
  const long long total = (long long)MM * PD_;
  if (i >= total) return;
  int col = (int)(i % PD_);
  long long row = i / PD_;
  int b = (int)(row / LL);
  int tokrem = (int)(row % LL);
  int t = tokrem / NP_;
  int hw = tokrem % NP_;
  int hg = hw / 8, wg = hw % 8;
  int c = col / 64;
  int r = col % 64;
  int ph = r / 8, pw = r % 8;
  size_t src = ((((size_t)b * TT + t) * CC + c) * RES_ + (hg * 8 + ph)) * RES_ +
               (wg * 8 + pw);
  A[i] = (bf16)fr[src];
}

// tokens [B*L, PD] -> out [B,T,C,RES,RES]
__global__ void k_unpatchify(const float* __restrict__ tok,
                             float* __restrict__ out) {
  long long i = (long long)blockIdx.x * blockDim.x + threadIdx.x;
  const long long total = (long long)BB * TT * CC * RES_ * RES_;
  if (i >= total) return;
  int xw = (int)(i % RES_);
  int y = (int)((i / RES_) % RES_);
  int c = (int)((i / ((long long)RES_ * RES_)) % CC);
  int t = (int)((i / ((long long)RES_ * RES_ * CC)) % TT);
  int b = (int)(i / ((long long)RES_ * RES_ * CC * TT));
  int hg = y >> 3, ph = y & 7, wg = xw >> 3, pw = xw & 7;
  size_t row = (size_t)b * LL + t * NP_ + hg * 8 + wg;
  int col = c * 64 + ph * 8 + pw;
  out[i] = tok[row * PD_ + col];
}

// RMSNorm over N=512: fp32 in -> optional bf16 out, optional fp32 out
__global__ void k_rmsnorm(const float* __restrict__ x,
                          const float* __restrict__ scale,
                          bf16* __restrict__ obf, float* __restrict__ of32,
                          int N) {
  __shared__ float red[256];
  const int row = blockIdx.x;
  const float* xr = x + (size_t)row * N;
  float s = 0.f;
  for (int i = threadIdx.x; i < N; i += blockDim.x) {
    float v = xr[i];
    s += v * v;
  }
  red[threadIdx.x] = s;
  __syncthreads();
  for (int st = 128; st > 0; st >>= 1) {
    if ((int)threadIdx.x < st) red[threadIdx.x] += red[threadIdx.x + st];
    __syncthreads();
  }
  const float inv = rsqrtf(red[0] / (float)N + EPS_);
  for (int i = threadIdx.x; i < N; i += blockDim.x) {
    float v = xr[i] * inv * scale[i];
    if (obf)  obf[(size_t)row * N + i] = (bf16)v;
    if (of32) of32[(size_t)row * N + i] = v;
  }
}

// SwiGLU-combine: silu(g1)*u1 + silu(g2)*u2 -> bf16 [M, 682]
__global__ void k_swiglu(const float* __restrict__ g,
                         const float* __restrict__ u, bf16* __restrict__ o) {
  long long i = (long long)blockIdx.x * blockDim.x + threadIdx.x;
  const long long total = (long long)MM * HALF_;
  if (i >= total) return;
  int j = (int)(i % HALF_);
  long long m = i / HALF_;
  size_t r = (size_t)m * INNER_;
  float g1 = g[r + j], g2 = g[r + j + HALF_];
  float u1 = u[r + j], u2 = u[r + j + HALF_];
  float s1 = g1 / (1.f + __expf(-g1));
  float s2 = g2 / (1.f + __expf(-g2));
  o[(size_t)m * HALF_ + j] = (bf16)(s1 * u1 + s2 * u2);
}

// Per-head RMSNorm + RoPE on q/k, passthrough v.
// qkv fp32 [B,L,3,NH,HD] -> q,k,v bf16 [B,NH,L,HD]
__global__ void k_rope(const float* __restrict__ qkv,
                       const float* __restrict__ q_scale,
                       const float* __restrict__ k_scale,
                       bf16* __restrict__ q, bf16* __restrict__ k,
                       bf16* __restrict__ v) {
  __shared__ float rq[64], rk[64], sq[64], sk[64];
  int bid = blockIdx.x;  // over B*NH*L
  const int ll = bid % LL; bid /= LL;
  const int hh = bid % NHD; bid /= NHD;
  const int b = bid;
  const int d = threadIdx.x;  // 0..63

  const size_t base = ((((size_t)b * LL + ll) * 3 + 0) * NHD + hh) * HD_;
  const float qv = qkv[base + d];
  const float kv = qkv[base + (size_t)NHD * HD_ + d];
  const float vv = qkv[base + 2 * (size_t)NHD * HD_ + d];

  rq[d] = qv * qv;
  rk[d] = kv * kv;
  __syncthreads();
  for (int st = 32; st > 0; st >>= 1) {
    if (d < st) { rq[d] += rq[d + st]; rk[d] += rk[d + st]; }
    __syncthreads();
  }
  const float qi = qv * rsqrtf(rq[0] / (float)HD_ + EPS_) * q_scale[d];
  const float ki = kv * rsqrtf(rk[0] / (float)HD_ + EPS_) * k_scale[d];
  sq[d] = qi;
  sk[d] = ki;
  __syncthreads();

  // first RD dims: temporal rope (pos = frame idx), rest: spatial (pos = patch)
  const int seg = (d < RD_) ? 0 : 1;
  const int dd = seg ? (d - RD_) : d;
  const int pos = seg ? (ll % NP_) : (ll / NP_);
  const int j = dd >> 1;
  const float invf = __expf(-((float)(2 * j) / (float)RD_) * 9.210340372f); // ln 1e4
  const float ang = (float)pos * invf;
  const float cs = __cosf(ang), sn = __sinf(ang);
  const int partner = (d & 1) ? (d - 1) : (d + 1);
  const float sign = (d & 1) ? 1.f : -1.f;
  const float qo = qi * cs + sign * sq[partner] * sn;
  const float ko = ki * cs + sign * sk[partner] * sn;

  const size_t o = (((size_t)b * NHD + hh) * LL + ll) * HD_ + d;
  q[o] = (bf16)qo;
  k[o] = (bf16)ko;
  v[o] = (bf16)vv;
}

// ---------------------------------------------------------------------------
// Generic bf16 GEMM: C[M,N] = A[M,K] @ W[K,N] (+ R) ; fp32 accumulate.
// 256 threads = 8 waves; tile 128x128; each wave 32x64 (2x4 WMMA frags).
// A-tile staged with async global->LDS copies.
// ---------------------------------------------------------------------------
__global__ void k_gemm(const bf16* __restrict__ A, const bf16* __restrict__ Wt,
                       float* __restrict__ Cc, const float* __restrict__ Rres,
                       int Mm, int Nn, int Kk) {
  __shared__ __align__(16) bf16 As[128][32];
  __shared__ __align__(16) bf16 Bt[128][32];  // [n][k] transposed stage
  const int tid = threadIdx.x;
  const int m0 = blockIdx.y * 128;
  const int n0 = blockIdx.x * 128;
  const int lane = tid & 31, w = tid >> 5;
  const int wm = w & 3, wn = w >> 2;
  const int lm = lane & 15, lh = lane >> 4;

  v8f acc[2][4];
#pragma unroll
  for (int mi = 0; mi < 2; ++mi)
#pragma unroll
    for (int ni = 0; ni < 4; ++ni)
#pragma unroll
      for (int jj = 0; jj < 8; ++jj) acc[mi][ni][jj] = 0.f;

  const bool kvec = (Kk & 31) == 0;
  const int nk = (Kk + 31) >> 5;
  for (int kt = 0; kt < nk; ++kt) {
    const int k0 = kt << 5;
    if (kvec) {
      // 128x32 A tile: 16B async copies, memory -> LDS without touching VGPRs
      for (int i = tid; i < 512; i += 256) {
        int r = i >> 2, c = (i & 3) << 3;
        int gm = m0 + r;
        if (gm < Mm) async_copy16(A + (size_t)gm * Kk + k0 + c, &As[r][c]);
      }
    } else {
      for (int i = tid; i < 4096; i += 256) {
        int r = i >> 5, c = i & 31;
        int gm = m0 + r, gk = k0 + c;
        As[r][c] = (gm < Mm && gk < Kk) ? A[(size_t)gm * Kk + gk] : (bf16)0.f;
      }
    }
    for (int i = tid; i < 4096; i += 256) {
      int c = i >> 7, n = i & 127;  // c = k within tile (coalesced global read)
      int gk = k0 + c, gn = n0 + n;
      Bt[n][c] = (gk < Kk && gn < Nn) ? Wt[(size_t)gk * Nn + gn] : (bf16)0.f;
    }
    // prefetch next k-tile while this one is staged
    if (k0 + 32 < Kk) {
      __builtin_prefetch(A + (size_t)(m0 + (tid >> 1)) * Kk + k0 + 32, 0, 1);
      __builtin_prefetch(Wt + (size_t)(k0 + 32 + (tid >> 7)) * Nn + n0 + (tid & 127), 0, 1);
    }
    wait_async_lds();
    __syncthreads();

    v16bf a0 = frag_a(&As[wm * 32 + lm][0], lh);
    v16bf a1 = frag_a(&As[wm * 32 + 16 + lm][0], lh);
#pragma unroll
    for (int ni = 0; ni < 4; ++ni) {
      v16bf bfr = frag_b(&Bt[wn * 64 + ni * 16 + lm][0], lh);
      acc[0][ni] = wmma_bf16(a0, bfr, acc[0][ni]);
      acc[1][ni] = wmma_bf16(a1, bfr, acc[1][ni]);
    }
    __syncthreads();
  }

#pragma unroll
  for (int mi = 0; mi < 2; ++mi)
#pragma unroll
    for (int ni = 0; ni < 4; ++ni)
#pragma unroll
      for (int vv = 0; vv < 8; ++vv) {
        int gm = m0 + wm * 32 + mi * 16 + vv + lh * 8;
        int gn = n0 + wn * 64 + ni * 16 + lm;
        if (gm < Mm && gn < Nn) {
          float val = acc[mi][ni][vv];
          if (Rres) val += Rres[(size_t)gm * Nn + gn];
          Cc[(size_t)gm * Nn + gn] = val;
        }
      }
}

// ---------------------------------------------------------------------------
// Flash attention with block-causal mask. 64-token tiles == frames, so the
// mask is exact at tile granularity (q-frame t attends k-frames 0..t fully).
// One workgroup per (b, head, q-frame); 4 waves; 128 threads.
// ---------------------------------------------------------------------------
__global__ void k_attn(const bf16* __restrict__ qb, const bf16* __restrict__ kb,
                       const bf16* __restrict__ vb, bf16* __restrict__ ob) {
  __shared__ __align__(16) bf16 Qs[64][64];
  __shared__ __align__(16) bf16 Ks[64][64];
  __shared__ __align__(16) bf16 Vt[64][64];  // [hd][token]
  __shared__ __align__(16) bf16 Pb[64][64];
  __shared__ float Sb[64][64];
  __shared__ float m_s[64], l_s[64], f_s[64];

  const int tid = threadIdx.x;
  const int lane = tid & 31, w = tid >> 5;  // 4 waves; wave w owns q rows 16w..
  const int lm = lane & 15, lh = lane >> 4;
  int bid = blockIdx.x;
  const int qt = bid % TT; bid /= TT;
  const int hh = bid % NHD; bid /= NHD;
  const int b = bid;
  const size_t headbase = ((size_t)b * NHD + hh) * LL;

  for (int i = tid; i < 64 * 8; i += 128) {
    int r = i >> 3, c = (i & 7) << 3;
    async_copy16(qb + (headbase + qt * 64 + r) * HD_ + c, &Qs[r][c]);
  }
  if (tid < 64) { m_s[tid] = -3.0e38f; l_s[tid] = 0.f; }
  v8f of[4];
#pragma unroll
  for (int ni = 0; ni < 4; ++ni)
#pragma unroll
    for (int jj = 0; jj < 8; ++jj) of[ni][jj] = 0.f;

  for (int kt = 0; kt <= qt; ++kt) {
    for (int i = tid; i < 64 * 8; i += 128) {
      int r = i >> 3, c = (i & 7) << 3;
      async_copy16(kb + (headbase + kt * 64 + r) * HD_ + c, &Ks[r][c]);
    }
    for (int i = tid; i < 4096; i += 128) {
      int r = i >> 6, c = i & 63;
      Vt[c][r] = vb[(headbase + kt * 64 + r) * HD_ + c];
    }
    wait_async_lds();
    __syncthreads();

    // S = Q @ K^T * (1/sqrt(HD))
    v8f sacc[4];
#pragma unroll
    for (int ni = 0; ni < 4; ++ni)
#pragma unroll
      for (int jj = 0; jj < 8; ++jj) sacc[ni][jj] = 0.f;
#pragma unroll
    for (int ks = 0; ks < 2; ++ks) {
      v16bf aq = frag_a(&Qs[w * 16 + lm][ks * 32], lh);
#pragma unroll
      for (int ni = 0; ni < 4; ++ni) {
        v16bf bk = frag_b(&Ks[ni * 16 + lm][ks * 32], lh);
        sacc[ni] = wmma_bf16(aq, bk, sacc[ni]);
      }
    }
#pragma unroll
    for (int ni = 0; ni < 4; ++ni)
#pragma unroll
      for (int vv = 0; vv < 8; ++vv)
        Sb[w * 16 + vv + lh * 8][ni * 16 + lm] = sacc[ni][vv] * 0.125f;
    __syncthreads();

    // online softmax, one thread per q-row
    if (tid < 64) {
      float mo = m_s[tid];
      float mx = mo;
      for (int j = 0; j < 64; ++j) mx = fmaxf(mx, Sb[tid][j]);
      float alpha = __expf(mo - mx);
      float sum = 0.f;
      for (int j = 0; j < 64; ++j) {
        float pv = __expf(Sb[tid][j] - mx);
        Pb[tid][j] = (bf16)pv;
        sum += pv;
      }
      m_s[tid] = mx;
      l_s[tid] = l_s[tid] * alpha + sum;
      f_s[tid] = alpha;
    }
    __syncthreads();

    // rescale running O and accumulate P @ V
#pragma unroll
    for (int ni = 0; ni < 4; ++ni)
#pragma unroll
      for (int vv = 0; vv < 8; ++vv) of[ni][vv] *= f_s[w * 16 + vv + lh * 8];
#pragma unroll
    for (int ks = 0; ks < 2; ++ks) {
      v16bf ap = frag_a(&Pb[w * 16 + lm][ks * 32], lh);
#pragma unroll
      for (int ni = 0; ni < 4; ++ni) {
        v16bf bv = frag_b(&Vt[ni * 16 + lm][ks * 32], lh);
        of[ni] = wmma_bf16(ap, bv, of[ni]);
      }
    }
    __syncthreads();
  }

  // normalize and write o as bf16 [B, L, D] (head-major inside D)
#pragma unroll
  for (int ni = 0; ni < 4; ++ni)
#pragma unroll
    for (int vv = 0; vv < 8; ++vv) {
      int m = w * 16 + vv + lh * 8;
      int n = ni * 16 + lm;
      int token = qt * 64 + m;
      ob[(((size_t)b * LL + token) * DD) + hh * HD_ + n] =
          (bf16)(of[ni][vv] / l_s[m]);
    }
}

// ---------------------------------------------------------------------------
// Host orchestration
// ---------------------------------------------------------------------------
extern "C" void kernel_launch(void* const* d_in, const int* in_sizes, int n_in,
                              void* d_out, int out_size, void* d_ws,
                              size_t ws_size, hipStream_t stream) {
  (void)in_sizes; (void)n_in; (void)out_size; (void)ws_size;

  const float* frames     = (const float*)d_in[0];
  const float* w_embed_f  = (const float*)d_in[1];
  const float* embed_norm = (const float*)d_in[2];
  const float* out_norm   = (const float*)d_in[3];
  const float* w_head_f   = (const float*)d_in[4];
  const float *lw_qkv[NLAY], *lw_out[NLAY], *l_qs[NLAY], *l_ks[NLAY],
      *l_n1[NLAY], *l_n2[NLAY], *lw_gate[NLAY], *lw_up[NLAY], *lw_down[NLAY];
  for (int l = 0; l < NLAY; ++l) {
    const int base = 5 + l * 9;
    lw_qkv[l]  = (const float*)d_in[base + 0];
    lw_out[l]  = (const float*)d_in[base + 1];
    l_qs[l]    = (const float*)d_in[base + 2];
    l_ks[l]    = (const float*)d_in[base + 3];
    l_n1[l]    = (const float*)d_in[base + 4];
    l_n2[l]    = (const float*)d_in[base + 5];
    lw_gate[l] = (const float*)d_in[base + 6];
    lw_up[l]   = (const float*)d_in[base + 7];
    lw_down[l] = (const float*)d_in[base + 8];
  }

  char* p = (char*)d_ws;
  auto alloc = [&](size_t bytes) -> void* {
    void* r = (void*)p;
    p += (bytes + 255) & ~(size_t)255;
    return r;
  };
  float* x    = (float*)alloc((size_t)MM * DD * 4);
  float* xt   = (float*)alloc((size_t)MM * DD * 4);
  bf16*  hb   = (bf16*)alloc((size_t)MM * DD * 2);
  float* qkvb = (float*)alloc((size_t)MM * 3 * DD * 4);
  bf16*  qbb  = (bf16*)alloc((size_t)MM * DD * 2);
  bf16*  kbb  = (bf16*)alloc((size_t)MM * DD * 2);
  bf16*  vbb  = (bf16*)alloc((size_t)MM * DD * 2);
  bf16*  obb  = (bf16*)alloc((size_t)MM * DD * 2);
  float* gbuf = (float*)alloc((size_t)MM * INNER_ * 4);
  float* ubuf = (float*)alloc((size_t)MM * INNER_ * 4);
  bf16*  swb  = (bf16*)alloc((size_t)MM * HALF_ * 2);
  bf16*  apat = (bf16*)alloc((size_t)MM * PD_ * 2);
  float* tok  = (float*)alloc((size_t)MM * PD_ * 4);
  bf16*  wembed = (bf16*)alloc((size_t)PD_ * DD * 2);
  bf16*  whead  = (bf16*)alloc((size_t)DD * PD_ * 2);
  bf16 *wqkv[NLAY], *wout[NLAY], *wgate[NLAY], *wup[NLAY], *wdown[NLAY];
  for (int l = 0; l < NLAY; ++l) {
    wqkv[l]  = (bf16*)alloc((size_t)DD * 3 * DD * 2);
    wout[l]  = (bf16*)alloc((size_t)DD * DD * 2);
    wgate[l] = (bf16*)alloc((size_t)DD * INNER_ * 2);
    wup[l]   = (bf16*)alloc((size_t)DD * INNER_ * 2);
    wdown[l] = (bf16*)alloc((size_t)HALF_ * DD * 2);
  }

  auto conv = [&](const float* s, bf16* d, size_t n) {
    k_f32_to_bf16<<<dim3((unsigned)((n + 255) / 256)), 256, 0, stream>>>(
        s, d, (int)n);
  };
  conv(w_embed_f, wembed, (size_t)PD_ * DD);
  conv(w_head_f, whead, (size_t)DD * PD_);
  for (int l = 0; l < NLAY; ++l) {
    conv(lw_qkv[l], wqkv[l], (size_t)DD * 3 * DD);
    conv(lw_out[l], wout[l], (size_t)DD * DD);
    conv(lw_gate[l], wgate[l], (size_t)DD * INNER_);
    conv(lw_up[l], wup[l], (size_t)DD * INNER_);
    conv(lw_down[l], wdown[l], (size_t)HALF_ * DD);
  }

  const dim3 blk256(256);
  // patchify + embedding
  {
    size_t n = (size_t)MM * PD_;
    k_patchify<<<dim3((unsigned)((n + 255) / 256)), blk256, 0, stream>>>(frames,
                                                                         apat);
  }
  k_gemm<<<dim3((DD + 127) / 128, (MM + 127) / 128), blk256, 0, stream>>>(
      apat, wembed, xt, nullptr, MM, DD, PD_);
  k_rmsnorm<<<MM, blk256, 0, stream>>>(xt, embed_norm, nullptr, x, DD);

  for (int l = 0; l < NLAY; ++l) {
    k_rmsnorm<<<MM, blk256, 0, stream>>>(x, l_n1[l], hb, nullptr, DD);
    k_gemm<<<dim3((3 * DD + 127) / 128, (MM + 127) / 128), blk256, 0, stream>>>(
        hb, wqkv[l], qkvb, nullptr, MM, 3 * DD, DD);
    k_rope<<<BB * NHD * LL, 64, 0, stream>>>(qkvb, l_qs[l], l_ks[l], qbb, kbb,
                                             vbb);
    k_attn<<<BB * NHD * TT, 128, 0, stream>>>(qbb, kbb, vbb, obb);
    k_gemm<<<dim3((DD + 127) / 128, (MM + 127) / 128), blk256, 0, stream>>>(
        obb, wout[l], x, x, MM, DD, DD);  // fused residual
    k_rmsnorm<<<MM, blk256, 0, stream>>>(x, l_n2[l], hb, nullptr, DD);
    k_gemm<<<dim3((INNER_ + 127) / 128, (MM + 127) / 128), blk256, 0, stream>>>(
        hb, wgate[l], gbuf, nullptr, MM, INNER_, DD);
    k_gemm<<<dim3((INNER_ + 127) / 128, (MM + 127) / 128), blk256, 0, stream>>>(
        hb, wup[l], ubuf, nullptr, MM, INNER_, DD);
    {
      size_t n = (size_t)MM * HALF_;
      k_swiglu<<<dim3((unsigned)((n + 255) / 256)), blk256, 0, stream>>>(
          gbuf, ubuf, swb);
    }
    k_gemm<<<dim3((DD + 127) / 128, (MM + 127) / 128), blk256, 0, stream>>>(
        swb, wdown[l], x, x, MM, DD, HALF_);  // fused residual, K=682 tail
  }

  k_rmsnorm<<<MM, blk256, 0, stream>>>(x, out_norm, hb, nullptr, DD);
  k_gemm<<<dim3((PD_ + 127) / 128, (MM + 127) / 128), blk256, 0, stream>>>(
      hb, whead, tok, nullptr, MM, PD_, DD);
  {
    size_t n = (size_t)BB * TT * CC * RES_ * RES_;
    k_unpatchify<<<dim3((unsigned)((n + 255) / 256)), blk256, 0, stream>>>(
        tok, (float*)d_out);
  }
}
